// RegionProposalNetwork_53102975647947
// MI455X (gfx1250) — matrix operand order repaired and checked by reading, not compiled, not run on hardware
//
#include <hip/hip_runtime.h>
#include <hip/hip_bf16.h>

typedef __attribute__((ext_vector_type(16))) _Float16 v16h;
typedef __attribute__((ext_vector_type(8)))  _Float16 v8h;
typedef __attribute__((ext_vector_type(8)))  float    v8f;
typedef unsigned int       u32;
typedef unsigned long long u64;

#define NIMG    16
#define CIN     256
#define COUT    256
#define HW      64
#define NPIX    (NIMG * HW * HW)          // 65536
#define KTOT    (CIN * 9)                 // 2304
#define NKT     (KTOT / 32)               // 72 K-steps
#define NANCH   9
#define NSC     36864                     // 64*64*9 scores per image
#define NSORT   65536                     // padded, power of two
#define PRE_N   1000
#define NMS_THR 0.7f
#define MIN_SZ  1.0f
#define BBOX_CLIP 4.135166556742356f      // log(1000/16)

#define XP_H    66                        // padded height (64 + halo)
#define XP_W    72                        // padded row (64 + halo + align slack)

// CDNA5 async global->LDS (ASYNCcnt) if this toolchain exposes it;
// otherwise fall back to the register-staged path (also used by the
// host-side compile pass, where the builtin does not exist).
// Builtin signature (from clang diagnostic): param 1 is
// 'int __vector(4) addrspace(1)*', param 2 the LDS-side pointer.
#if __has_builtin(__builtin_amdgcn_global_load_async_to_lds_b128)
#define HAVE_ASYNC_LDS 1
typedef int v4i_ __attribute__((vector_size(16)));
typedef __attribute__((address_space(1))) v4i_ ga_v4i;   // global int4
typedef __attribute__((address_space(3))) v4i_ ls_v4i;   // LDS int4
#else
#define HAVE_ASYNC_LDS 0
#endif

__device__ __forceinline__ void async_wait0() {
#if HAVE_ASYNC_LDS
#if __has_builtin(__builtin_amdgcn_s_wait_asynccnt)
    __builtin_amdgcn_s_wait_asynccnt(0);
#else
    asm volatile("s_wait_asynccnt 0x0" ::: "memory");
#endif
#endif
}

// ---------------- workspace layout (bytes; every size is 256B-multiple) ----
#define WS_HIDDEN   0ull                                     // 32 MB f16 NHWC
#define WS_KEYS     (WS_HIDDEN + (u64)NPIX * 256 * 2)        //  8 MB u64
#define WS_DELTAS   (WS_KEYS   + (u64)NIMG * NSORT * 8)      //  9 MB f32
#define WS_BOXES    (WS_DELTAS + (u64)NIMG * NSC * 4 * 4)
#define WS_VALID    (WS_BOXES  + (u64)NIMG * PRE_N * 4 * 4)
#define WS_XPAD     (WS_VALID  + (u64)NIMG * PRE_N * 4)      // 38.9 MB f16 padded NCHW
#define WS_WT       (WS_XPAD   + (u64)NIMG * CIN * XP_H * XP_W * 2) // 1.1 MB f16

__device__ __forceinline__ u32 f32_sortable(float s) {
    u32 u = __float_as_uint(s);
    return (u & 0x80000000u) ? ~u : (u | 0x80000000u);
}

// =====================================================================
// Prep A: features f32 NCHW -> f16 zero-padded [b][ci][66][72]
// =====================================================================
__global__ __launch_bounds__(256)
void rpn_prep_features(const float* __restrict__ X, _Float16* __restrict__ Xp) {
    const u64 q = (u64)blockIdx.x * 256 + threadIdx.x;      // quad index
    const u64 total_q = (u64)NIMG * CIN * XP_H * XP_W / 4;
    if (q >= total_q) return;
    const int xx0 = (int)((q * 4) % XP_W);
    const u64 row = (q * 4) / XP_W;
    const int yy  = (int)(row % XP_H);
    const u64 bc  = row / XP_H;                             // b*256+ci

    _Float16 v[4];
#pragma unroll
    for (int i = 0; i < 4; ++i) {
        const int xx = xx0 + i;
        float f = 0.0f;
        if (yy >= 1 && yy <= HW && xx >= 1 && xx <= HW)
            f = X[bc * (HW * HW) + (u64)(yy - 1) * HW + (xx - 1)];
        v[i] = (_Float16)f;
    }
    *reinterpret_cast<uint2*>(Xp + q * 4) = *reinterpret_cast<const uint2*>(v);
}

// =====================================================================
// Prep B: conv weights f32 [co][ci*9+r] -> f16 permuted [co][r*256+ci]
// =====================================================================
__global__ __launch_bounds__(256)
void rpn_prep_weights(const float* __restrict__ Wc, _Float16* __restrict__ Wt) {
    const int t = blockIdx.x * 256 + threadIdx.x;
    if (t >= COUT * KTOT) return;
    const int co = t / KTOT;
    const int kp = t - co * KTOT;
    const int r  = kp >> 8;
    const int ci = kp & 255;
    Wt[t] = (_Float16)Wc[(size_t)co * KTOT + ci * 9 + r];
}

// =====================================================================
// Kernel 1: 3x3 conv as implicit GEMM, permuted-K, double-buffered LDS,
// software-pipelined.  Block: 256 thr (8 waves), tile 64(M) x 128(N).
// B tile staged with GLOBAL_LOAD_ASYNC_TO_LDS_B128 when available.
// =====================================================================
__global__ __launch_bounds__(256)
void rpn_conv3x3_wmma(const _Float16* __restrict__ Xp,
                      const _Float16* __restrict__ Wt,
                      const float* __restrict__ Bc,
                      _Float16* __restrict__ hidden) {
    __shared__ _Float16 As[2][64][40];    // [buf][m=x][k=ci_off], 16B-aligned rows
    __shared__ _Float16 Bs[2][128][40];   // [buf][n][k]

    const int tid  = threadIdx.x;
    const int wave = tid >> 5;
    const int lane = tid & 31;
    const int half = lane >> 4;
    const int l15  = lane & 15;
    const int mg   = wave & 3;
    const int ng   = wave >> 2;

    const int m0  = blockIdx.x * 64;           // one image row
    const int co0 = blockIdx.y * 128;
    const int bB  = m0 >> 12;                  // image (block-uniform)
    const int yB  = (m0 & 4095) >> 6;          // row    (block-uniform)

    // A loader: ci_off = tid&31, x chunk of 8 at (tid>>5)*8
    const int aci = tid & 31;
    const int ax0 = (tid >> 5) * 8;
    // B loader: n = tid>>1, 16-wide k-half = (tid&1)*16
    const int bn  = tid >> 1;
    const int bkh = (tid & 1) * 16;

    v8f acc[4];
#pragma unroll
    for (int j = 0; j < 4; ++j) acc[j] = (v8f){};

    _Float16 areg[8];
#if !HAVE_ASYNC_LDS
    uint4 breg0, breg1;
#endif

    auto fetchA = [&](int kt) {
        const int r   = kt >> 3;               // filter tap, fixed per K-step
        const int ky  = r / 3;
        const int kx  = r - ky * 3;
        const int ci0 = (kt & 7) * 32;
        const _Float16* arow =
            Xp + ((u64)(bB * CIN + ci0 + aci) * XP_H + (yB + ky)) * XP_W + kx + ax0;
#pragma unroll
        for (int i = 0; i < 8; ++i) areg[i] = arow[i];      // contiguous, merges to b128
    };
    auto stageA = [&](int buf) {
#pragma unroll
        for (int i = 0; i < 8; ++i) As[buf][ax0 + i][aci] = areg[i];
    };
    auto fetchB = [&](int kt, int buf) {
        const _Float16* wp = Wt + (u64)(co0 + bn) * KTOT + kt * 32 + bkh;
#if HAVE_ASYNC_LDS
        // 32B per thread, global -> LDS DMA, no VGPR round-trip (ASYNCcnt)
        __builtin_amdgcn_global_load_async_to_lds_b128(
            (ga_v4i*)wp, (ls_v4i*)&Bs[buf][bn][bkh], 0, 0);
        __builtin_amdgcn_global_load_async_to_lds_b128(
            (ga_v4i*)wp, (ls_v4i*)&Bs[buf][bn][bkh], 16, 0);
#else
        const uint4* w4 = reinterpret_cast<const uint4*>(wp);
        breg0 = w4[0];
        breg1 = w4[1];
        (void)buf;
#endif
    };
    auto stageB = [&](int buf) {
#if !HAVE_ASYNC_LDS
        uint4* bp = reinterpret_cast<uint4*>(&Bs[buf][bn][bkh]);
        bp[0] = breg0;
        bp[1] = breg1;
#else
        (void)buf;
#endif
    };
    auto compute = [&](int buf) {
        v16h af;                               // A 16x32, contiguous k = 16*half + i (as B)
        const _Float16* ap = &As[buf][16 * mg + l15][0];
#pragma unroll
        for (int i = 0; i < 8; ++i) {
            af[i]     = ap[16 * half + i];
            af[8 + i] = ap[16 * half + 8 + i];
        }
        v16h bf[4];                            // preload all B frags: one dscnt
#pragma unroll
        for (int j = 0; j < 4; ++j) {          // wait, 4 back-to-back WMMAs
            const _Float16* bp = &Bs[buf][64 * ng + 16 * j + l15][0];
#pragma unroll
            for (int i = 0; i < 16; ++i) bf[j][i] = bp[16 * half + i];
        }
#pragma unroll
        for (int j = 0; j < 4; ++j)
            acc[j] = __builtin_amdgcn_wmma_f32_16x16x32_f16(
                false, af, false, bf[j], (short)0, acc[j], false, false);
    };

    fetchA(0);
    fetchB(0, 0);
    stageA(0);
    stageB(0);
    async_wait0();                             // B tile 0 resident before barrier
    for (int kt = 0; kt < NKT; ++kt) {
        __syncthreads();                       // buf[kt&1] ready; buf[(kt+1)&1] free
        const int nxt = kt + 1;
        if (nxt < NKT) {
            fetchA(nxt);                       // issue global / async DMA early
            fetchB(nxt, nxt & 1);
        }
        compute(kt & 1);                       // WMMAs overlap the transfers
        if (nxt < NKT) {
            stageA(nxt & 1);
            stageB(nxt & 1);
        }
        async_wait0();                         // DMA done before next barrier
    }

    // ---- epilogue: bias + ReLU, store NHWC f16 ----
#pragma unroll
    for (int j = 0; j < 4; ++j) {
        const int ch = co0 + 64 * ng + 16 * j + l15;
        const float bias = Bc[ch];
#pragma unroll
        for (int r = 0; r < 8; ++r) {
            const int pp = m0 + 16 * mg + r + 8 * half;
            float v = acc[j][r] + bias;
            v = fmaxf(v, 0.0f);
            hidden[(u64)pp * COUT + ch] = (_Float16)v;
        }
    }
}

// =====================================================================
// Kernel 2: fused 1x1 score+loc GEMM on WMMA. N = 48 (9 score, 36 loc,
// 3 pad), K = 256. Emits sort keys (u64) and deltas (f32).
// =====================================================================
__global__ __launch_bounds__(128)
void rpn_heads_wmma(const _Float16* __restrict__ hidden,
                    const float* __restrict__ Ws,
                    const float* __restrict__ Bs_b,
                    const float* __restrict__ Wl,
                    const float* __restrict__ Bl_b,
                    u64* __restrict__ keys,
                    float* __restrict__ deltas) {
    __shared__ _Float16 Bw[48][264];

    const int tid  = threadIdx.x;
    const int wave = tid >> 5;
    const int lane = tid & 31;
    const int half = lane >> 4;
    const int l15  = lane & 15;
    const int p0   = blockIdx.x * 64;

    for (int e = tid; e < 48 * 256; e += 128) {
        const int n = e >> 8;
        const int k = e & 255;
        float w = 0.0f;
        if (n < 9)       w = Ws[n * 256 + k];
        else if (n < 45) w = Wl[(n - 9) * 256 + k];
        Bw[n][k] = (_Float16)w;
    }
    __syncthreads();

    v8f acc[3];
#pragma unroll
    for (int j = 0; j < 3; ++j) acc[j] = (v8f){};

    const _Float16* hrow = hidden + (u64)(p0 + 16 * wave + l15) * COUT;
#pragma unroll
    for (int kt = 0; kt < 256 / 32; ++kt) {
        v16h af;
        const _Float16* ap = hrow + kt * 32;
#pragma unroll
        for (int i = 0; i < 8; ++i) {
            af[i]     = ap[16 * half + i];
            af[8 + i] = ap[16 * half + 8 + i];
        }
#pragma unroll
        for (int j = 0; j < 3; ++j) {
            v16h bf;
            const _Float16* bp = &Bw[16 * j + l15][kt * 32];
#pragma unroll
            for (int i = 0; i < 16; ++i) bf[i] = bp[16 * half + i];
            acc[j] = __builtin_amdgcn_wmma_f32_16x16x32_f16(
                false, af, false, bf, (short)0, acc[j], false, false);
        }
    }

#pragma unroll
    for (int j = 0; j < 3; ++j) {
        const int c = 16 * j + l15;
        float bias = 0.0f;
        if (c < 9)       bias = Bs_b[c];
        else if (c < 45) bias = Bl_b[c - 9];
#pragma unroll
        for (int r = 0; r < 8; ++r) {
            const int pp = p0 + 16 * wave + r + 8 * half;
            const int b  = pp >> 12;
            const int pi = pp & 4095;
            const float v = acc[j][r] + bias;
            if (c < 9) {
                const u32 idx = (u32)(pi * NANCH + c);
                const u64 key = ((u64)(~f32_sortable(v)) << 32) | (u64)idx;
                keys[(u64)b * NSORT + idx] = key;
            } else if (c < 45) {
                const int d = c - 9;
                deltas[(((u64)b * NSC) + (u64)pi * NANCH + (d >> 2)) * 4 + (d & 3)] = v;
            }
        }
    }
}

// =====================================================================
// Kernel 3: per-image bitonic sort of 65536 u64 keys (ascending =>
// descending score, index tie-break). One block per image, 1024 thr.
// =====================================================================
__global__ __launch_bounds__(1024)
void rpn_sort(u64* __restrict__ keys) {
    const int tid = threadIdx.x;
    u64* k = keys + (u64)blockIdx.x * NSORT;

    for (int i = NSC + tid; i < NSORT; i += 1024)
        k[i] = ~0ull;
    __syncthreads();

    for (u32 kk = 2; kk <= NSORT; kk <<= 1) {
        for (u32 j = kk >> 1; j > 0; j >>= 1) {
            for (int t = tid; t < NSORT; t += 1024) {
                const u32 partner = (u32)t ^ j;
                if (partner > (u32)t) {
                    const u64 a = k[t];
                    const u64 b = k[partner];
                    const bool asc = ((u32)t & kk) == 0;
                    if (asc ? (a > b) : (a < b)) {
                        k[t]       = b;
                        k[partner] = a;
                    }
                }
            }
            __syncthreads();
        }
    }
}

// =====================================================================
// Kernel 4: decode top-1000: anchors on the fly, delta decode, clip,
// min-size validity. grid = (16 images, 4 chunks of 256).
// =====================================================================
__global__ __launch_bounds__(256)
void rpn_decode(const u64* __restrict__ keys,
                const float* __restrict__ deltas,
                const int* __restrict__ img_h_p,
                const int* __restrict__ img_w_p,
                float* __restrict__ boxesT,
                int* __restrict__ validT) {
    const int b = blockIdx.x;
    const int i = blockIdx.y * 256 + threadIdx.x;
    if (i >= PRE_N) return;

    const float img_hf = (float)(*img_h_p);
    const float img_wf = (float)(*img_w_p);
    const float sy = img_hf / (float)HW;
    const float sx = img_wf / (float)HW;

    const u32 idx   = (u32)(keys[(u64)b * NSORT + i] & 0xFFFFFFFFull);
    const int pixel = (int)(idx / NANCH);
    const int a     = (int)(idx - (u32)pixel * NANCH);
    const int y     = pixel >> 6;
    const int x     = pixel & 63;

    const float ratios[3] = {0.5f, 1.0f, 2.0f};
    const float sizes_[3] = {128.0f, 256.0f, 512.0f};
    const float hr = __fsqrt_rn(ratios[a / 3]);
    const float wa = (1.0f / hr) * sizes_[a % 3];
    const float ha = hr * sizes_[a % 3];
    const float cx = (float)x * sx;
    const float cy = (float)y * sy;

    const u64 dbase = (((u64)b * NSC) + (u64)idx) * 4;
    const float dx = deltas[dbase + 0];
    const float dy = deltas[dbase + 1];
    const float dw = fminf(deltas[dbase + 2], BBOX_CLIP);
    const float dh = fminf(deltas[dbase + 3], BBOX_CLIP);

    const float pcx = dx * wa + cx;
    const float pcy = dy * ha + cy;
    const float pw  = __expf(dw) * wa;
    const float ph  = __expf(dh) * ha;

    float x1 = fminf(fmaxf(pcx - 0.5f * pw, 0.0f), img_wf);
    float y1 = fminf(fmaxf(pcy - 0.5f * ph, 0.0f), img_hf);
    float x2 = fminf(fmaxf(pcx + 0.5f * pw, 0.0f), img_wf);
    float y2 = fminf(fmaxf(pcy + 0.5f * ph, 0.0f), img_hf);

    float* bp = boxesT + ((u64)b * PRE_N + i) * 4;
    bp[0] = x1; bp[1] = y1; bp[2] = x2; bp[3] = y2;
    validT[(u64)b * PRE_N + i] =
        ((x2 - x1) >= MIN_SZ && (y2 - y1) >= MIN_SZ) ? 1 : 0;
}

// =====================================================================
// Kernel 5: greedy NMS (LDS-resident) + ordered compaction + zero pad.
// One block per image, 1024 threads.
// =====================================================================
__global__ __launch_bounds__(1024)
void rpn_nms(const float* __restrict__ boxesT,
             const int* __restrict__ validT,
             float* __restrict__ out) {
    __shared__ float bx[PRE_N][4];
    __shared__ float area[PRE_N];
    __shared__ int   keep[PRE_N];
    __shared__ int   valid_s[PRE_N];
    __shared__ int   sc[1024];
    __shared__ int   supp;

    const int b   = blockIdx.x;
    const int tid = threadIdx.x;
    const float* bp = boxesT + (u64)b * PRE_N * 4;

    for (int i = tid; i < PRE_N; i += 1024) {
        bx[i][0] = bp[i * 4 + 0];
        bx[i][1] = bp[i * 4 + 1];
        bx[i][2] = bp[i * 4 + 2];
        bx[i][3] = bp[i * 4 + 3];
        area[i]  = (bx[i][2] - bx[i][0]) * (bx[i][3] - bx[i][1]);
        keep[i]  = 0;
        valid_s[i] = validT[(u64)b * PRE_N + i];
    }
    __syncthreads();

    for (int i = 0; i < PRE_N; ++i) {
        if (tid == 0) supp = 0;
        __syncthreads();
        const int j = tid;
        if (j < i && keep[j]) {
            const float lx = fmaxf(bx[i][0], bx[j][0]);
            const float ly = fmaxf(bx[i][1], bx[j][1]);
            const float rx = fminf(bx[i][2], bx[j][2]);
            const float ry = fminf(bx[i][3], bx[j][3]);
            const float iw = fmaxf(rx - lx, 0.0f);
            const float ih = fmaxf(ry - ly, 0.0f);
            const float inter = iw * ih;
            const float iou = inter / (area[i] + area[j] - inter + 1e-9f);
            if (iou > NMS_THR) atomicOr(&supp, 1);
        }
        __syncthreads();
        if (tid == 0) keep[i] = (valid_s[i] && !supp) ? 1 : 0;
        __syncthreads();
    }

    sc[tid] = (tid < PRE_N) ? keep[tid] : 0;
    __syncthreads();
    for (int off = 1; off < 1024; off <<= 1) {
        const int mine = sc[tid];
        const int add  = (tid >= off) ? sc[tid - off] : 0;
        __syncthreads();
        sc[tid] = mine + add;
        __syncthreads();
    }
    const int total = sc[1023];

    float* op = out + (u64)b * PRE_N * 4;
    if (tid < PRE_N && keep[tid]) {
        const int pos = sc[tid] - keep[tid];
        op[pos * 4 + 0] = bx[tid][0];
        op[pos * 4 + 1] = bx[tid][1];
        op[pos * 4 + 2] = bx[tid][2];
        op[pos * 4 + 3] = bx[tid][3];
    }
    __syncthreads();
    for (int r = tid; r < PRE_N; r += 1024) {
        if (r >= total) {
            op[r * 4 + 0] = 0.0f; op[r * 4 + 1] = 0.0f;
            op[r * 4 + 2] = 0.0f; op[r * 4 + 3] = 0.0f;
        }
    }
}

// =====================================================================
extern "C" void kernel_launch(void* const* d_in, const int* in_sizes, int n_in,
                              void* d_out, int out_size, void* d_ws, size_t ws_size,
                              hipStream_t stream) {
    const float* features = (const float*)d_in[0];
    const float* conv_w   = (const float*)d_in[1];
    const float* conv_b   = (const float*)d_in[2];
    const float* score_w  = (const float*)d_in[3];
    const float* score_b  = (const float*)d_in[4];
    const float* loc_w    = (const float*)d_in[5];
    const float* loc_b    = (const float*)d_in[6];
    const int*   img_h    = (const int*)d_in[7];
    const int*   img_w    = (const int*)d_in[8];

    char* ws = (char*)d_ws;
    _Float16* hidden = (_Float16*)(ws + WS_HIDDEN);
    u64*      keys   = (u64*)(ws + WS_KEYS);
    float*    deltas = (float*)(ws + WS_DELTAS);
    float*    boxesT = (float*)(ws + WS_BOXES);
    int*      validT = (int*)(ws + WS_VALID);
    _Float16* xpad   = (_Float16*)(ws + WS_XPAD);
    _Float16* wt     = (_Float16*)(ws + WS_WT);
    float*    out    = (float*)d_out;

    // 0) one-shot preps: padded f16 features, permuted f16 weights
    {
        const u64 total_q = (u64)NIMG * CIN * XP_H * XP_W / 4;
        rpn_prep_features<<<(u32)((total_q + 255) / 256), 256, 0, stream>>>(features, xpad);
        rpn_prep_weights<<<(COUT * KTOT + 255) / 256, 256, 0, stream>>>(conv_w, wt);
    }

    // 1) 3x3 conv + ReLU -> hidden (NHWC f16)
    dim3 g1(NPIX / 64, COUT / 128);
    rpn_conv3x3_wmma<<<g1, 256, 0, stream>>>(xpad, wt, conv_b, hidden);

    // 2) fused score/loc 1x1 GEMM -> keys + deltas
    rpn_heads_wmma<<<NPIX / 64, 128, 0, stream>>>(hidden, score_w, score_b,
                                                  loc_w, loc_b, keys, deltas);

    // 3) per-image sort (top-k ordering)
    rpn_sort<<<NIMG, 1024, 0, stream>>>(keys);

    // 4) decode + clip + validity for top-1000
    dim3 g4(NIMG, (PRE_N + 255) / 256);
    rpn_decode<<<g4, 256, 0, stream>>>(keys, deltas, img_h, img_w, boxesT, validT);

    // 5) NMS + ordered compaction -> d_out
    rpn_nms<<<NIMG, 1024, 0, stream>>>(boxesT, validT, out);
}
